// GAT_41824391528577
// MI455X (gfx1250) — matrix operand order repaired and checked by reading, not compiled, not run on hardware
//
#include <hip/hip_runtime.h>
#include <hip/hip_bf16.h>
#include <math.h>

typedef __attribute__((ext_vector_type(16))) __bf16 v16bf;
typedef __attribute__((ext_vector_type(2)))  __bf16 v2bf;
typedef __attribute__((ext_vector_type(8)))  float  v8f;
typedef __attribute__((ext_vector_type(4)))  int    v4i;
typedef __attribute__((ext_vector_type(4)))  float  v4f;

#define N_NODES 4096
#define IN_F    128
#define NHID    256
#define NHEAD   4
#define FPH     64
#define OUTF    128
#define NEG_SLOPE 0.2f
#define LOG2E   1.4426950408889634f

__device__ __forceinline__ unsigned short f2bf(float f) {
    union { float f; unsigned u; } x; x.f = f;
    unsigned r = x.u + 0x7fffu + ((x.u >> 16) & 1u);   // round-to-nearest-even
    return (unsigned short)(r >> 16);
}

__device__ __forceinline__ float fast_exp2(float z) {
#if __has_builtin(__builtin_amdgcn_exp2f)
    return __builtin_amdgcn_exp2f(z);     // bare v_exp_f32
#else
    return exp2f(z);
#endif
}

__device__ __forceinline__ float fast_rcp(float z) {
#if __has_builtin(__builtin_amdgcn_rcpf)
    return __builtin_amdgcn_rcpf(z);      // single v_rcp_f32
#else
    return 1.0f / z;
#endif
}

// ---------------- Kernel A: g = x @ W1 ; also store gT (bf16, [256][4096]) ----
__global__ __launch_bounds__(256) void gat_proj_kernel(
    const float* __restrict__ x, const float* __restrict__ W1,
    float* __restrict__ g, unsigned short* __restrict__ gT)
{
    __shared__ float xs[16][IN_F];
    const int n0  = blockIdx.x << 4;
    const int tid = threadIdx.x;

    for (int t = tid; t < 16 * IN_F; t += 256)
        xs[t >> 7][t & 127] = x[(size_t)(n0 + (t >> 7)) * IN_F + (t & 127)];
    __syncthreads();

    const int c = tid;                  // output column 0..255
    float acc[16];
#pragma unroll
    for (int r = 0; r < 16; ++r) acc[r] = 0.f;

    for (int k = 0; k < IN_F; ++k) {
        float w = W1[(size_t)k * NHID + c];
#pragma unroll
        for (int r = 0; r < 16; ++r) acc[r] = fmaf(xs[r][k], w, acc[r]);
    }
#pragma unroll
    for (int r = 0; r < 16; ++r) {
        g[(size_t)(n0 + r) * NHID + c] = acc[r];
        gT[(size_t)c * N_NODES + n0 + r] = f2bf(acc[r]);
    }
}

// ---- Kernel B: el/er = (g . attn_{l,r}) * log2(e)   ([4][4096] layout) ------
// Pre-scaling by log2e lets the softmax use raw v_exp_f32 (exp2); valid since
// leaky_relu commutes with positive scaling and softmax is base-invariant then.
__global__ __launch_bounds__(256) void gat_attnvec_kernel(
    const float* __restrict__ g, const float* __restrict__ al,
    const float* __restrict__ ar, float* __restrict__ elT, float* __restrict__ erT)
{
    const int idx = blockIdx.x * 256 + threadIdx.x;   // 16384 = 4096*4
    const int n = idx >> 2;
    const int h = idx & 3;
    const float* gr = g + (size_t)n * NHID + h * FPH;
    float sl = 0.f, sr = 0.f;
#pragma unroll 8
    for (int f = 0; f < FPH; ++f) {
        float v = gr[f];
        sl = fmaf(v, al[f], sl);
        sr = fmaf(v, ar[f], sr);
    }
    elT[h * N_NODES + n] = sl * LOG2E;
    erT[h * N_NODES + n] = sr * LOG2E;
}

// ---------------- Kernel C: fused masked-softmax aggregation + ELU + W_out ----
// 256 blocks (16-row i-tiles) x 128 threads (4 waves; wave = head).
// Softmax denominator computed by a 5th WMMA against an all-ones B fragment:
// D[M][N] = sum_k P[M,k] -> each lane holds exactly the row sums its C rows need.
__global__ __launch_bounds__(128) void gat_agg_kernel(
    const int* __restrict__ adj, const unsigned short* __restrict__ gT,
    const float* __restrict__ elT, const float* __restrict__ erT,
    const float* __restrict__ W_out, const float* __restrict__ b_out,
    float* __restrict__ out)
{
    __shared__ float res[16][NHID];       // attn_res tile after ELU

    const int tid  = threadIdx.x;
    const int lane = tid & 31;
    const int h    = tid >> 5;            // head: one wave per head
    const int l15  = lane & 15;           // A: row M ; B/D: col N
    const int grp  = lane >> 4;
    const int i0   = blockIdx.x << 4;
    const int kb   = grp << 4;            // A-fragment K base (0 or 16)
    const int cb   = grp << 4;            // B-fragment K base (0 or 16)

    const float el_i = elT[(h << 12) + i0 + l15];
    const float* erh = erT + (h << 12);
    const int*   adjrow = adj + (size_t)(i0 + l15) * N_NODES;
    const unsigned short* gTh = gT + (size_t)((h << 6) + l15) * N_NODES;

    // all-ones B fragment (bf16 1.0 = 0x3F80) for the denominator WMMA
    union { v16bf v; unsigned short u[16]; } Bones;
#pragma unroll
    for (int t = 0; t < 16; ++t) Bones.u[t] = 0x3F80u;

    v8f acc[5];
#pragma unroll
    for (int ft = 0; ft < 5; ++ft) acc[ft] = (v8f){};

    const float NEG_INF = -__builtin_inff();

    for (int jt = 0; jt < N_NODES; jt += 32) {
        // adjacency (streamed once -> nontemporal) and er for this lane's 16 K's
        union { v4i v[4]; int s[16]; } aj;
        aj.v[0] = __builtin_nontemporal_load((const v4i*)(adjrow + jt + kb));
        aj.v[1] = __builtin_nontemporal_load((const v4i*)(adjrow + jt + kb + 4));
        aj.v[2] = __builtin_nontemporal_load((const v4i*)(adjrow + jt + kb + 8));
        aj.v[3] = __builtin_nontemporal_load((const v4i*)(adjrow + jt + kb + 12));
        union { v4f v[4]; float s[16]; } er;
        er.v[0] = *(const v4f*)(erh + jt + kb);
        er.v[1] = *(const v4f*)(erh + jt + kb + 4);
        er.v[2] = *(const v4f*)(erh + jt + kb + 8);
        er.v[3] = *(const v4f*)(erh + jt + kb + 12);

        // A fragment: w = exp2( adj ? leaky_relu(el+er) : -inf )
        float w[16];
#pragma unroll
        for (int t = 0; t < 16; ++t) {
            float z = el_i + er.s[t];
            z = fmaxf(z, NEG_SLOPE * z);
            z = aj.s[t] ? z : NEG_INF;
            w[t] = fast_exp2(z);
        }
        union { v16bf v; v2bf p[8]; __bf16 b[16]; } A;
#if __has_builtin(__builtin_amdgcn_cvt_pk_bf16_f32)
#pragma unroll
        for (int t = 0; t < 8; ++t)
            A.p[t] = __builtin_amdgcn_cvt_pk_bf16_f32(w[2 * t], w[2 * t + 1]);
#else
#pragma unroll
        for (int t = 0; t < 16; ++t) A.b[t] = (__bf16)w[t];
#endif

        // B fragments: gT row = h*64 + ft*16 + l15, K-contiguous 16 bf16 (32B)
        const unsigned short* gcol = gTh + jt + cb;
        v16bf B0 = *(const v16bf*)(gcol);
        v16bf B1 = *(const v16bf*)(gcol + 16 * N_NODES);
        v16bf B2 = *(const v16bf*)(gcol + 32 * N_NODES);
        v16bf B3 = *(const v16bf*)(gcol + 48 * N_NODES);

        acc[0] = __builtin_amdgcn_wmma_f32_16x16x32_bf16(false, A.v, false, B0,
                                                         (short)0, acc[0], false, false);
        acc[1] = __builtin_amdgcn_wmma_f32_16x16x32_bf16(false, A.v, false, B1,
                                                         (short)0, acc[1], false, false);
        acc[2] = __builtin_amdgcn_wmma_f32_16x16x32_bf16(false, A.v, false, B2,
                                                         (short)0, acc[2], false, false);
        acc[3] = __builtin_amdgcn_wmma_f32_16x16x32_bf16(false, A.v, false, B3,
                                                         (short)0, acc[3], false, false);
        acc[4] = __builtin_amdgcn_wmma_f32_16x16x32_bf16(false, A.v, false, Bones.v,
                                                         (short)0, acc[4], false, false);
    }

    // acc[4][r] = denominator of row (r + 8*grp): exactly the rows this lane owns
    float rinv[8];
#pragma unroll
    for (int r = 0; r < 8; ++r) rinv[r] = fast_rcp(acc[4][r]);

    // divide, ELU, stash to LDS   (C/D layout: VGPR r -> row r + 8*grp, col l15)
#pragma unroll
    for (int ft = 0; ft < 4; ++ft) {
#pragma unroll
        for (int r = 0; r < 8; ++r) {
            float v = acc[ft][r] * rinv[r];
            v = v > 0.f ? v : (__expf(v) - 1.f);
            res[r + (grp << 3)][(h << 6) + (ft << 4) + l15] = v;
        }
    }
    __syncthreads();

    // epilogue: out[i0+r][o] = elu(res) @ W_out + b_out ; thread = output col o
    const int o = tid;                    // 0..127
    float s[16];
#pragma unroll
    for (int r = 0; r < 16; ++r) s[r] = b_out[o];
    for (int c = 0; c < NHID; ++c) {
        float w = W_out[(size_t)c * OUTF + o];
#pragma unroll
        for (int r = 0; r < 16; ++r) s[r] = fmaf(res[r][c], w, s[r]);
    }
#pragma unroll
    for (int r = 0; r < 16; ++r)
        out[(size_t)(i0 + r) * OUTF + o] = s[r];
}

extern "C" void kernel_launch(void* const* d_in, const int* in_sizes, int n_in,
                              void* d_out, int out_size, void* d_ws, size_t ws_size,
                              hipStream_t stream) {
    const float* x     = (const float*)d_in[0];
    const int*   adj   = (const int*)  d_in[1];
    const float* W1    = (const float*)d_in[2];
    const float* al    = (const float*)d_in[3];
    const float* ar    = (const float*)d_in[4];
    const float* W_out = (const float*)d_in[5];
    const float* b_out = (const float*)d_in[6];
    float* out = (float*)d_out;

    // workspace layout
    float*          g   = (float*)d_ws;                                    // 4 MB
    unsigned short* gT  = (unsigned short*)((char*)d_ws + (size_t)N_NODES * NHID * 4); // 2 MB
    float*          elT = (float*)((char*)d_ws + (size_t)6 * 1024 * 1024); // 64 KB
    float*          erT = elT + NHEAD * N_NODES;                           // 64 KB

    gat_proj_kernel   <<<N_NODES / 16, 256, 0, stream>>>(x, W1, g, gT);
    gat_attnvec_kernel<<<(N_NODES * NHEAD) / 256, 256, 0, stream>>>(g, al, ar, elT, erT);
    gat_agg_kernel    <<<N_NODES / 16, 128, 0, stream>>>(adj, gT, elT, erT, W_out, b_out, out);
}
